// Refiner_24644522344904
// MI455X (gfx1250) — matrix operand order repaired and checked by reading, not compiled, not run on hardware
//
#include <hip/hip_runtime.h>
#include <hip/hip_bf16.h>

// ---------------------------------------------------------------------------
// MI455X (gfx1250). Compute-bound patch CNN -> bf16 WMMA implicit GEMM.
// Per-patch pipeline in LDS (2 x 24x24x96 bf16 ping-pong ~221KB of 320KB/WGP).
// This round: fully-unrolled conv steps (immediate-offset loads, compiler can
// pipeline), 2 M-tiles per B fragment (half weight traffic, interleaved WMMA
// chains hide D->C hazard NOPs), nt-major weight layout for WGP$ locality.
// ---------------------------------------------------------------------------

typedef __attribute__((ext_vector_type(16))) __bf16 v16bf;
typedef __attribute__((ext_vector_type(8)))  float  v8f;
typedef __attribute__((ext_vector_type(4)))  unsigned int v4u;

union Frag { v4u u4[2]; v16bf bf; };

__device__ __forceinline__ unsigned short f2bf(float x) {
  unsigned u = __float_as_uint(x);
  return (unsigned short)((u + 0x7FFFu + ((u >> 16) & 1u)) >> 16);  // RNE
}
__device__ __forceinline__ float bf2f(unsigned short h) {
  return __uint_as_float(((unsigned)h) << 16);
}

// ---------------------------------------------------------------------------
// Kernel 1: fold BN, re-tile weights into bf16 WMMA B-blocks, nt-major:
//   block index within layer = (nt*9 + tap)*KS + ks   (KS=1 for L0, else 3)
// Per 32x16 block (512 bf16): lane l holds 16 contiguous bf16 =
//   K in [(l>=16)*16, +16) for fixed N = l&15  -> two b128 loads per lane.
// ---------------------------------------------------------------------------
__global__ void prep_kernel(const float* __restrict__ w0, const float* __restrict__ wpre,
                            const float* __restrict__ wpost,
                            const float* __restrict__ bn0, const float* __restrict__ bnpre,
                            const float* __restrict__ bnpost,
                            unsigned short* __restrict__ wout, float* __restrict__ bnout) {
  int gid = blockIdx.x * blockDim.x + threadIdx.x;
  if (gid < 714240) {
    int layer, tap, nt, ks, s;
    if (gid < 23040) {
      layer = 0; int blk = gid >> 9; s = gid & 511;
      nt = blk / 9; tap = blk % 9; ks = 0;
    } else {
      int r = gid - 23040; layer = 1 + r / 69120; r %= 69120;
      int blk = r >> 9; s = r & 511;
      nt = blk / 27; int rem = blk % 27; tap = rem / 3; ks = rem % 3;
    }
    int lane = s >> 4, j = s & 15;
    int n = lane & 15;
    int klocal = ((lane >> 4) << 4) + j;   // 0..31
    int cout = nt * 16 + n;
    int cin  = ks * 32 + klocal;
    float v = 0.0f;
    if (layer == 0)      { if (cout < 80 && cin < 31) v = w0[(cout * 31 + cin) * 9 + tap]; }
    else if (layer <= 3) { if (cout < 80 && cin < 80) v = wpre[(((layer - 1) * 80 + cout) * 80 + cin) * 9 + tap]; }
    else                 { if (cout < 80 && cin < 80) v = wpost[(((layer - 4) * 80 + cout) * 80 + cin) * 9 + tap]; }
    wout[gid] = f2bf(v);
  }
  if (gid < 880) {  // 11 BN sets x 80 ch -> scale/bias
    int set = gid / 80, ch = gid % 80;
    const float* p;
    if (set == 0)      p = bn0;
    else if (set <= 3) p = bnpre + (set - 1) * 320;
    else               p = bnpost + (set - 4) * 320;
    float g = p[ch], bb = p[80 + ch], m = p[160 + ch], vv = p[240 + ch];
    float sc = g * rsqrtf(vv + 1e-5f);
    bnout[set * 80 + ch]       = sc;
    bnout[880 + set * 80 + ch] = bb - m * sc;
  }
}

// ---------------------------------------------------------------------------
// Kernel 2: per-batch top-512 of 4096, jax.lax.top_k stable-tie semantics.
// ---------------------------------------------------------------------------
__global__ __launch_bounds__(256) void topk_kernel(const float* __restrict__ err,
                                                   int* __restrict__ list,
                                                   float* __restrict__ ref_out) {
  __shared__ float sv[4096];
  __shared__ float so[4096];
  __shared__ int scnt[257];
  __shared__ int sg[257];
  const int tid = threadIdx.x;
  const int b = blockIdx.x;
  const float* e = err + b * 4096;
  for (int i = tid; i < 4096; i += 256) { float v = e[i]; sv[i] = v; so[i] = v; }
  __syncthreads();
  for (int k = 2; k <= 4096; k <<= 1) {            // bitonic ascending sort
    for (int j = k >> 1; j > 0; j >>= 1) {
      for (int i = tid; i < 4096; i += 256) {
        int p = i ^ j;
        if (p > i) {
          bool up = ((i & k) == 0);
          float a = sv[i], c = sv[p];
          if ((a > c) == up) { sv[i] = c; sv[p] = a; }
        }
      }
      __syncthreads();
    }
  }
  const float thr = sv[4096 - 512];
  const int base = tid * 16;
  int cg = 0, ce = 0;
  for (int q = 0; q < 16; ++q) { float v = so[base + q]; cg += (v > thr); ce += (v == thr); }
  scnt[tid] = ce; sg[tid] = cg;
  __syncthreads();
  if (tid == 0) {
    int aE = 0, aG = 0;
    for (int t = 0; t < 256; ++t) {
      int tE = scnt[t]; scnt[t] = aE; aE += tE;
      int tG = sg[t];   sg[t]   = aG; aG += tG;
    }
    scnt[256] = aE; sg[256] = aG;
  }
  __syncthreads();
  const int need = 512 - sg[256];
  int eRank = scnt[tid];
  int cs = 0, selmask = 0;
  for (int q = 0; q < 16; ++q) {
    float v = so[base + q];
    bool sel = (v > thr) || ((v == thr) && (eRank < need));
    if (v == thr) eRank++;
    if (sel) { selmask |= (1 << q); cs++; }
  }
  __syncthreads();
  scnt[tid] = cs;
  __syncthreads();
  if (tid == 0) { int a = 0; for (int t = 0; t < 256; ++t) { int x = scnt[t]; scnt[t] = a; a += x; } }
  __syncthreads();
  int slot = scnt[tid];
  for (int q = 0; q < 16; ++q) {
    int idx = base + q;
    bool sel = (selmask >> q) & 1;
    if (sel) { list[b * 512 + slot] = (b << 12) | idx; slot++; }
    ref_out[b * 4096 + idx] = (sel && so[idx] > 0.0f) ? 1.0f : 0.0f;
  }
}

// ---------------------------------------------------------------------------
// Kernel 3: copy pha -> out.
// ---------------------------------------------------------------------------
__global__ void copy_pha_kernel(const float4* __restrict__ s, float4* __restrict__ d) {
  int i = blockIdx.x * 256 + threadIdx.x;
  d[i] = s[i];
}

// ---------------------------------------------------------------------------
// WMMA conv layer, specialized on KS (K-steps of 32 per tap) and residual.
// Fully unrolled 9*KS steps: all LDS/global loads use immediate offsets from
// per-tile base pointers -> compiler clusters loads ahead of WMMAs.
// Two M-tiles per iteration share one B fragment; the two accumulator chains
// interleave so no WMMA depends on the immediately preceding one.
// ---------------------------------------------------------------------------
__device__ __forceinline__ void store_tile(unsigned short* __restrict__ outb,
                                           const unsigned short* __restrict__ inb,
                                           const v8f& acc, int mt, int SS, int S, int org,
                                           int chO, float sc, float bi, int nhalf, bool resid) {
#pragma unroll
  for (int r = 0; r < 8; ++r) {
    const int mr = mt * 16 + nhalf * 8 + r;
    if (mr < SS) {
      const int ry = mr / S, rx = mr % S;
      const int fp = ((org + ry) * 24 + (org + rx)) * 96 + chO;
      float v = acc[r] * sc + bi;
      v = v > 0.f ? v : 0.f;
      if (resid) v += bf2f(inb[fp]);
      outb[fp] = f2bf(v);
    }
  }
}

template <int KS, bool RESID>
__device__ __forceinline__ void conv_layer(const unsigned short* __restrict__ inb,
                                           unsigned short* __restrict__ outb,
                                           const unsigned short* __restrict__ wl,
                                           const float* __restrict__ bnS,
                                           const float* __restrict__ bnB,
                                           int S, int org, int wave, int lane) {
  const int ncol = lane & 15, nhalf = lane >> 4;
  const int SS = S * S;
  const int Mtiles = (SS + 15) >> 4;
  const int Mpairs = (Mtiles + 1) >> 1;
  const int tiles = Mpairs * 5;
  for (int t = wave; t < tiles; t += 8) {
    const int nt = t / Mpairs;          // nt-major: 8 waves share one 27KB B window
    const int mtp = t % Mpairs;
    const int mt0 = mtp * 2, mt1 = mt0 + 1;
    int m0 = mt0 * 16 + ncol; if (m0 >= SS) m0 = SS - 1;
    int m1 = mt1 * 16 + ncol; if (m1 >= SS) m1 = SS - 1;
    const int oy0 = m0 / S, ox0 = m0 % S;
    const int oy1 = m1 / S, ox1 = m1 % S;
    // per-lane base pointers; all step offsets are compile-time immediates
    const unsigned short* a0 = inb + (((org - 1 + oy0) * 24 + (org - 1 + ox0)) * 96 + nhalf * 16);
    const unsigned short* a1 = inb + (((org - 1 + oy1) * 24 + (org - 1 + ox1)) * 96 + nhalf * 16);
    const unsigned short* bp = wl + ((nt * 9 * KS) << 9) + (lane << 4);
    v8f acc0 = {0.f, 0.f, 0.f, 0.f, 0.f, 0.f, 0.f, 0.f};
    v8f acc1 = {0.f, 0.f, 0.f, 0.f, 0.f, 0.f, 0.f, 0.f};
#pragma unroll
    for (int tap = 0; tap < 9; ++tap) {
      const int pofs = ((tap / 3) * 24 + (tap % 3)) * 96;
#pragma unroll
      for (int ks = 0; ks < KS; ++ks) {
        const int aofs = pofs + ks * 32;               // A: lane<16 K={0..15}
        const int bofs = (tap * KS + ks) << 9;         //    lane>=16 K={16..31}
        Frag A0, A1, B;
        A0.u4[0] = *(const v4u*)(a0 + aofs);
        A0.u4[1] = *(const v4u*)(a0 + aofs + 8);
        A1.u4[0] = *(const v4u*)(a1 + aofs);
        A1.u4[1] = *(const v4u*)(a1 + aofs + 8);
        B.u4[0]  = *(const v4u*)(bp + bofs);
        B.u4[1]  = *(const v4u*)(bp + bofs + 8);
        acc0 = __builtin_amdgcn_wmma_f32_16x16x32_bf16(
            false, A0.bf, false, B.bf, (short)0, acc0, false, false);
        acc1 = __builtin_amdgcn_wmma_f32_16x16x32_bf16(
            false, A1.bf, false, B.bf, (short)0, acc1, false, false);
      }
    }
    const int chO = nt * 16 + ncol;
    if (chO < 80) {
      const float sc = bnS[chO];
      const float bi = bnB[chO];
      store_tile(outb, inb, acc0, mt0, SS, S, org, chO, sc, bi, nhalf, RESID);
      store_tile(outb, inb, acc1, mt1, SS, S, org, chO, sc, bi, nhalf, RESID);
    }
  }
}

// ---------------------------------------------------------------------------
// Kernel 4: one workgroup (8 wave32) per patch.
// ---------------------------------------------------------------------------
__global__ __launch_bounds__(256) void refine_kernel(
    const int* __restrict__ list,
    const float* __restrict__ hid, const float* __restrict__ pha,
    const unsigned short* __restrict__ wbf, const float* __restrict__ bnw,
    const float* __restrict__ wlast, float* __restrict__ out) {
  __shared__ __align__(16) unsigned short sbuf[2][24 * 24 * 96];  // ~221 KB
  __shared__ float swl[720];
  const int tid = threadIdx.x;

  unsigned int* zb = (unsigned int*)sbuf;
  for (int e = tid; e < 24 * 24 * 96; e += 256) zb[e] = 0u;   // both bufs (as u32)
  for (int e = tid; e < 720; e += 256) swl[e] = wlast[e];
  __syncthreads();

  const int rec  = list[blockIdx.x];
  const int b    = rec >> 12, tile = rec & 4095;
  const int ti   = tile >> 6, tj = tile & 63;
  const int gy0  = ti * 16 - 4, gx0 = tj * 16 - 4;

  for (int c = 0; c < 31; ++c) {       // gather: ch0..29=hid, ch30=pha, rest 0
    const float* src = (c < 30) ? (hid + (size_t)(b * 30 + c) * (1024u * 1024u))
                                : (pha + (size_t)b * (1024u * 1024u));
    for (int p = tid; p < 576; p += 256) {
      int ry = p / 24, rx = p % 24;
      int gy = gy0 + ry, gx = gx0 + rx;
      if ((unsigned)gy < 1024u && (unsigned)gx < 1024u)
        sbuf[0][p * 96 + c] = f2bf(src[gy * 1024 + gx]);
    }
  }
  __syncthreads();

  const int oorgA[11] = {1, 2, 3, 4, 4, 4, 4, 4, 4, 4, 4};
  const int oszA[11]  = {22, 20, 18, 16, 16, 16, 16, 16, 16, 16, 16};
  const int wOffA[11] = {0, 23040, 92160, 161280, 230400, 299520, 368640,
                         437760, 506880, 576000, 645120};

  const int wave = tid >> 5, lane = tid & 31;

  for (int l = 0; l < 11; ++l) {
    const unsigned short* inb  = &sbuf[l & 1][0];
    unsigned short*       outb = &sbuf[(l + 1) & 1][0];
    const int S = oszA[l], org = oorgA[l];
    const unsigned short* wl = wbf + wOffA[l];
    const float* bnS = bnw + l * 80;
    const float* bnB = bnw + 880 + l * 80;

    if (l == 0)      conv_layer<1, false>(inb, outb, wl, bnS, bnB, S, org, wave, lane);
    else if (l < 4)  conv_layer<3, false>(inb, outb, wl, bnS, bnB, S, org, wave, lane);
    else             conv_layer<3, true >(inb, outb, wl, bnS, bnB, S, org, wave, lane);

    __syncthreads();
    if (l >= 3) {  // zero 1-px halo ring around 16x16 region for SAME convs
      unsigned int* ob = (unsigned int*)outb;
      for (int e = tid; e < 68 * 48; e += 256) {
        const int pxi = e / 48, u = e % 48;
        int py, px;
        if (pxi < 18)      { py = 3;               px = 3 + pxi; }
        else if (pxi < 36) { py = 20;              px = 3 + (pxi - 18); }
        else if (pxi < 52) { py = 4 + (pxi - 36);  px = 3; }
        else               { py = 4 + (pxi - 52);  px = 20; }
        ob[(py * 24 + px) * 48 + u] = 0u;
      }
      __syncthreads();
    }
  }

  // final conv 80->1, SAME on 16x16 (VALU; ~0.1% of FLOPs). L10 out = sbuf[1].
  const unsigned short* fin = &sbuf[1][0];
  const int oy = tid >> 4, ox = tid & 15;
  float s = 0.f;
  for (int tap = 0; tap < 9; ++tap) {
    const int iy = 3 + oy + tap / 3, ix = 3 + ox + tap % 3;
    const unsigned short* pp = &fin[(iy * 24 + ix) * 96];
    for (int c = 0; c < 80; ++c) s += bf2f(pp[c]) * swl[c * 9 + tap];
  }
  out[(size_t)b * (1024u * 1024u) + (size_t)(ti * 16 + oy) * 1024 + (tj * 16 + ox)] = s;
}

// ---------------------------------------------------------------------------
extern "C" void kernel_launch(void* const* d_in, const int* in_sizes, int n_in,
                              void* d_out, int out_size, void* d_ws, size_t ws_size,
                              hipStream_t stream) {
  const float* pha    = (const float*)d_in[0];
  const float* err    = (const float*)d_in[1];
  const float* hid    = (const float*)d_in[2];
  const float* w0     = (const float*)d_in[3];
  const float* bn0    = (const float*)d_in[4];
  const float* wpre   = (const float*)d_in[5];
  const float* bnpre  = (const float*)d_in[6];
  const float* wpost  = (const float*)d_in[7];
  const float* bnpost = (const float*)d_in[8];
  const float* wlast  = (const float*)d_in[9];
  float* out = (float*)d_out;
  char* ws = (char*)d_ws;
  int*            list = (int*)ws;                        // 1024 patch records
  float*          bnw  = (float*)(ws + 8192);             // 11x80 scale + bias
  unsigned short* wbf  = (unsigned short*)(ws + 16384);   // 714240 bf16 weights

  prep_kernel<<<(714240 + 255) / 256, 256, 0, stream>>>(w0, wpre, wpost, bn0, bnpre,
                                                        bnpost, wbf, bnw);
  topk_kernel<<<2, 256, 0, stream>>>(err, list, out + 2 * 1024 * 1024);
  copy_pha_kernel<<<2048, 256, 0, stream>>>((const float4*)pha, (float4*)out);
  refine_kernel<<<1024, 256, 0, stream>>>(list, hid, pha, wbf, bnw, wlast, out);
}
